// StructuralAwareContextModule_18700287607046
// MI455X (gfx1250) — compile-verified, matrix operand order checked by reading, not run on hardware
//
#include <hip/hip_runtime.h>
#include <math.h>

// ---------------------------------------------------------------------------
// StructuralAwareContextModule for MI455X (gfx1250, wave32, WMMA bf16)
//
//   B=1, S=2047 (padded to 2048), D=1024, H=16, HD=64
//   All matrix math on v_wmma_f32_16x16x32_bf16.
//   Flash-attention streaming softmax (scores never materialized in HBM).
//   S^T = K*Q^T trick: C-layout of S^T == A-layout of P for the P*V WMMA,
//   so no LDS transpose is needed anywhere in the attention inner loop.
// ---------------------------------------------------------------------------

#define S_LEN  2047
#define S_PAD  2048
#define DMODEL 1024
#define NHEAD  16
#define HDIM   64
#define KTILES (DMODEL / 32)  // 32 k-steps of 32 for the projections
#define NTILES (DMODEL / 16)  // 64 column tiles
#define MTILES (S_PAD / 16)   // 128 row tiles
#define RELN   (2 * S_PAD - 1) // 4095
#define NEDGE  1023

typedef __bf16         v16bf  __attribute__((ext_vector_type(16)));
typedef float          v8f    __attribute__((ext_vector_type(8)));
typedef unsigned short v8u16  __attribute__((ext_vector_type(8)));
typedef unsigned short v16u16 __attribute__((ext_vector_type(16)));

__device__ __forceinline__ unsigned short f2bf(float f) {
  union { float f; unsigned u; } x; x.f = f;
  unsigned r = x.u + 0x7FFFu + ((x.u >> 16) & 1u);   // round-to-nearest-even
  return (unsigned short)(r >> 16);
}

// Load one 16-bit WMMA fragment: two contiguous 16B chunks per lane.
// Works for A-frags (lane&15 = M row) and B-frags (lane&15 = N col) given the
// right base pointer; caller supplies p = row_base + (lane>>4)*8 element offset.
__device__ __forceinline__ v16bf load_frag(const unsigned short* p) {
  v8u16 lo = *(const v8u16*)p;
  v8u16 hi = *(const v8u16*)(p + 16);
  v16u16 w = __builtin_shufflevector(lo, hi, 0, 1, 2, 3, 4, 5, 6, 7,
                                     8, 9, 10, 11, 12, 13, 14, 15);
  return __builtin_bit_cast(v16bf, w);
}

__device__ __forceinline__ v8f wmma_bf16(v16bf a, v16bf b, v8f c) {
  return __builtin_amdgcn_wmma_f32_16x16x32_bf16(false, a, false, b,
                                                 (short)0, c, false, false);
}

// ---------------------------------------------------------------------------
// Packing kernels
// ---------------------------------------------------------------------------

// x (f32 [2047][1024]) -> bf16 [2048][1024], padded row zeroed
__global__ void pack_x_kernel(const float* __restrict__ x,
                              unsigned short* __restrict__ xb) {
  int idx = blockIdx.x * blockDim.x + threadIdx.x;   // 0 .. S_PAD*DMODEL-1
  int row = idx >> 10;
  xb[idx] = (row < S_LEN) ? f2bf(x[idx]) : (unsigned short)0;
}

// rel_emb [4095][16] -> posT [16][4095] (gathers along j become sequential)
__global__ void pack_pos_kernel(const float* __restrict__ rel_emb,
                                float* __restrict__ posT) {
  int idx = blockIdx.x * blockDim.x + threadIdx.x;
  if (idx >= NHEAD * RELN) return;
  int h = idx / RELN, r = idx % RELN;
  posT[idx] = rel_emb[r * NHEAD + h];
}

// W [N=1024][K=1024] f32 -> lane-contiguous bf16 B-fragments
// layout: [ntile(64)][ktile(32)][lane(32)][elem(16)]
__global__ void pack_w_kernel(const float* w0, const float* w1,
                              const float* w2, const float* w3,
                              unsigned short* b0, unsigned short* b1,
                              unsigned short* b2, unsigned short* b3) {
  const float* W; unsigned short* B;
  if (blockIdx.y == 0)      { W = w0; B = b0; }
  else if (blockIdx.y == 1) { W = w1; B = b1; }
  else if (blockIdx.y == 2) { W = w2; B = b2; }
  else                      { W = w3; B = b3; }
  int tid  = blockIdx.x * blockDim.x + threadIdx.x;  // 0..65535
  int lane = tid & 31;
  int kt   = (tid >> 5) & 31;
  int nt   = tid >> 10;
  int n    = nt * 16 + (lane & 15);
  int kb   = (lane >> 4) * 8;
  unsigned short* dst = B + (size_t)tid * 16;
#pragma unroll
  for (int e = 0; e < 8; ++e) {
    dst[e]     = f2bf(W[n * DMODEL + kt * 32 + kb + e]);        // K = kb..kb+7
    dst[e + 8] = f2bf(W[n * DMODEL + kt * 32 + 16 + kb + e]);   // K = 16+kb..
  }
}

// edge_vals[h][e] = dot(token_embeds[2e+1,:], W_edge[h,:]); one wave per dot.
__global__ void edge_kernel(const float* __restrict__ te,
                            const float* __restrict__ We,
                            float* __restrict__ ev) {
  int gwave = (blockIdx.x * blockDim.x + threadIdx.x) >> 5;
  int lane  = threadIdx.x & 31;
  if (gwave >= NHEAD * NEDGE) return;
  int h = gwave / NEDGE, e = gwave % NEDGE;
  const float* t = te + (size_t)(2 * e + 1) * DMODEL;
  const float* w = We + (size_t)h * DMODEL;
  float s = 0.0f;
  for (int c = lane; c < DMODEL; c += 32) s += t[c] * w[c];
#pragma unroll
  for (int m = 16; m; m >>= 1) s += __shfl_xor(s, m, 32);
  if (lane == 0) ev[h * NEDGE + e] = s;
}

// ---------------------------------------------------------------------------
// GEMM: D[2048][1024] = A[2048][1024] * Bpack^T (+bias). 4 waves/block, each
// wave computes one 16x64 tile (A-frag reused for 4 WMMAs per k-step).
// mode 0: Q  -> q[h][s][hd] bf16, pre-scaled by 1/sqrt(64)
// mode 1: K  -> k[h][s][hd] bf16
// mode 2: V  -> vt[h][hd][s] bf16 (transposed for the PV B-fragments)
// mode 3: out-> f32 d_out[2047][1024] with bias
// ---------------------------------------------------------------------------
__global__ void gemm_bf16_kernel(const unsigned short* __restrict__ A,
                                 const unsigned short* __restrict__ Bp,
                                 const float* __restrict__ bias,
                                 void* __restrict__ Out, int mode) {
  int lane = threadIdx.x & 31;
  int wave = threadIdx.x >> 5;
  int mt   = blockIdx.x;                       // 0..127
  int nt0  = (blockIdx.y * 4 + wave) * 4;      // 4 consecutive n-tiles
  int kb   = (lane >> 4) * 8;
  const unsigned short* aRow = A + (size_t)(mt * 16 + (lane & 15)) * DMODEL;
  const size_t bstride = (size_t)KTILES * 32 * 16;   // bytes-elems per ntile
  v8f acc0 = {}, acc1 = {}, acc2 = {}, acc3 = {};
  for (int kt = 0; kt < KTILES; ++kt) {
    const unsigned short* ap = aRow + kt * 32 + kb;
    if (kt + 1 < KTILES)
      __builtin_prefetch(aRow + (kt + 1) * 32 + kb, 0, 1); // global_prefetch_b8
    v16bf a = load_frag(ap);
    const unsigned short* bp =
        Bp + (((size_t)nt0 * KTILES + kt) * 32 + lane) * 16;
    v16bf b0 = __builtin_bit_cast(v16bf, *(const v16u16*)(bp));
    v16bf b1 = __builtin_bit_cast(v16bf, *(const v16u16*)(bp + bstride));
    v16bf b2 = __builtin_bit_cast(v16bf, *(const v16u16*)(bp + 2 * bstride));
    v16bf b3 = __builtin_bit_cast(v16bf, *(const v16u16*)(bp + 3 * bstride));
    acc0 = wmma_bf16(a, b0, acc0);
    acc1 = wmma_bf16(a, b1, acc1);
    acc2 = wmma_bf16(a, b2, acc2);
    acc3 = wmma_bf16(a, b3, acc3);
  }
  v8f accs[4] = {acc0, acc1, acc2, acc3};
  int lh = lane >> 4;
#pragma unroll
  for (int j = 0; j < 4; ++j) {
    int col = (nt0 + j) * 16 + (lane & 15);
    float bv = bias[col];
#pragma unroll
    for (int r = 0; r < 8; ++r) {
      int row = mt * 16 + r + 8 * lh;          // C layout: M = r + 8*(lane>>4)
      float v = accs[j][r] + bv;
      if (mode == 0) {
        ((unsigned short*)Out)[((size_t)((col >> 6) * S_PAD + row)) * HDIM +
                               (col & 63)] = f2bf(v * 0.125f);
      } else if (mode == 1) {
        ((unsigned short*)Out)[((size_t)((col >> 6) * S_PAD + row)) * HDIM +
                               (col & 63)] = f2bf(v);
      } else if (mode == 2) {
        ((unsigned short*)Out)[((size_t)((col >> 6) * HDIM + (col & 63))) *
                                   S_PAD + row] = f2bf(v);
      } else if (row < S_LEN) {
        ((float*)Out)[(size_t)row * DMODEL + col] = v;
      }
    }
  }
}

// ---------------------------------------------------------------------------
// Flash attention, one wave per (head, 16-query tile), 32 keys per block.
// S^T = K*Q^T so C-layout (lane = query, VGPR = key) feeds P*V directly.
// ---------------------------------------------------------------------------
__global__ void attention_kernel(const unsigned short* __restrict__ Q,
                                 const unsigned short* __restrict__ K,
                                 const unsigned short* __restrict__ VT,
                                 const float* __restrict__ posT,
                                 const float* __restrict__ ev,
                                 const int* __restrict__ mask,
                                 unsigned short* __restrict__ attnb) {
  int h    = blockIdx.x;                    // head
  int wave = threadIdx.x >> 5;
  int qt   = blockIdx.y * 4 + wave;         // query tile 0..127
  int lane = threadIdx.x & 31;
  int lh   = lane >> 4;
  int kb   = lh * 8;
  int qloc = lane & 15;
  int iq   = qt * 16 + qloc;                // the query this lane owns
  const unsigned short* Qh = Q + (size_t)h * S_PAD * HDIM;
  const unsigned short* Kh = K + (size_t)h * S_PAD * HDIM;
  const unsigned short* Vh = VT + (size_t)h * HDIM * S_PAD;
  const float* posR = posT + h * RELN + iq + (S_PAD - 1); // posR[-j]=bias(i,j)
  bool  has_edge = ((iq & 1) == 0) && (iq + 2 < S_LEN);   // edge at (i, i+2)
  float edge_v   = has_edge ? ev[h * NEDGE + (iq >> 1)] : 0.0f;

  // Q^T B-fragments, 2 k-steps of 32 over HD=64; reused for every key block.
  v16bf qf0 = load_frag(Qh + (size_t)iq * HDIM + kb);
  v16bf qf1 = load_frag(Qh + (size_t)iq * HDIM + 32 + kb);

  v8f o0 = {}, o1 = {}, o2 = {}, o3 = {};    // 16 queries x 64 hd accumulator
  float m_i = -__builtin_inff();
  float l_i = 0.0f;

  for (int kblk = 0; kblk < S_PAD; kblk += 32) {
    // S^T tiles: A = K rows (keys), B = Q^T
    v16bf ka0a = load_frag(Kh + (size_t)(kblk + qloc) * HDIM + kb);
    v16bf ka0b = load_frag(Kh + (size_t)(kblk + qloc) * HDIM + 32 + kb);
    v16bf ka1a = load_frag(Kh + (size_t)(kblk + 16 + qloc) * HDIM + kb);
    v16bf ka1b = load_frag(Kh + (size_t)(kblk + 16 + qloc) * HDIM + 32 + kb);
    v8f st0 = {}, st1 = {};
    st0 = wmma_bf16(ka0a, qf0, st0);
    st0 = wmma_bf16(ka0b, qf1, st0);
    st1 = wmma_bf16(ka1a, qf0, st1);
    st1 = wmma_bf16(ka1b, qf1, st1);

    // st*[r] = scores[key = kblk + (16*t) + r + 8*lh][query = iq]
    float s0[8], s1[8];
    float bm = -__builtin_inff();
#pragma unroll
    for (int r = 0; r < 8; ++r) {
      int j0 = kblk + 8 * lh + r;
      int j1 = j0 + 16;
      float v0 = -__builtin_inff();
      float v1 = -__builtin_inff();
      if (j0 < S_LEN && mask[j0] != 0) {
        v0 = st0[r] + posR[-j0];
        if (has_edge && j0 == iq + 2) v0 += edge_v;
      }
      if (j1 < S_LEN && mask[j1] != 0) {
        v1 = st1[r] + posR[-j1];
        if (has_edge && j1 == iq + 2) v1 += edge_v;
      }
      s0[r] = v0; s1[r] = v1;
      bm = fmaxf(bm, fmaxf(v0, v1));
    }
    // combine the two half-wave key ranges for this query
    bm = fmaxf(bm, __shfl_xor(bm, 16, 32));
    float m_new = fmaxf(m_i, bm);
    float corr  = __expf(m_i - m_new);       // first iter: exp(-inf) = 0
    float rsum  = 0.0f;
    v16u16 pw;
#pragma unroll
    for (int r = 0; r < 8; ++r) {
      float p0 = __expf(s0[r] - m_new);
      float p1 = __expf(s1[r] - m_new);
      rsum += p0 + p1;
      pw[r]     = f2bf(p0);                  // A-frag elems 0..7  = keys kb..
      pw[r + 8] = f2bf(p1);                  // A-frag elems 8..15 = keys 16+kb
    }
    rsum += __shfl_xor(rsum, 16, 32);
    l_i = l_i * corr + rsum;
    m_i = m_new;
    v16bf pf = __builtin_bit_cast(v16bf, pw);

    // rescale O rows: O row = r + 8*lh, corr lives on lane (r + 8*lh)
    float cr[8];
#pragma unroll
    for (int r = 0; r < 8; ++r) cr[r] = __shfl(corr, r + 8 * lh, 32);
#pragma unroll
    for (int r = 0; r < 8; ++r) {
      o0[r] *= cr[r]; o1[r] *= cr[r]; o2[r] *= cr[r]; o3[r] *= cr[r];
    }

    // O += P(16x32) * V(32x64); V B-frags from transposed vt[h][hd][s]
    o0 = wmma_bf16(pf, load_frag(Vh + (size_t)(0 * 16 + qloc) * S_PAD + kblk + kb), o0);
    o1 = wmma_bf16(pf, load_frag(Vh + (size_t)(1 * 16 + qloc) * S_PAD + kblk + kb), o1);
    o2 = wmma_bf16(pf, load_frag(Vh + (size_t)(2 * 16 + qloc) * S_PAD + kblk + kb), o2);
    o3 = wmma_bf16(pf, load_frag(Vh + (size_t)(3 * 16 + qloc) * S_PAD + kblk + kb), o3);
  }

  float linv = 1.0f / l_i;
  float lr[8];
#pragma unroll
  for (int r = 0; r < 8; ++r) lr[r] = __shfl(linv, r + 8 * lh, 32);
  v8f os[4] = {o0, o1, o2, o3};
#pragma unroll
  for (int nt = 0; nt < 4; ++nt) {
#pragma unroll
    for (int r = 0; r < 8; ++r) {
      int row = qt * 16 + r + 8 * lh;
      int col = h * HDIM + nt * 16 + qloc;
      float v = (row < S_LEN) ? os[nt][r] * lr[r] : 0.0f;
      attnb[(size_t)row * DMODEL + col] = f2bf(v);
    }
  }
}

// ---------------------------------------------------------------------------
extern "C" void kernel_launch(void* const* d_in, const int* in_sizes, int n_in,
                              void* d_out, int out_size, void* d_ws,
                              size_t ws_size, hipStream_t stream) {
  const float* x    = (const float*)d_in[0];
  /* d_in[1] token_types: unused by the reference */
  const float* te   = (const float*)d_in[2];
  const int*   mask = (const int*)d_in[3];
  const float* Wq   = (const float*)d_in[4];
  const float* bq   = (const float*)d_in[5];
  const float* Wk   = (const float*)d_in[6];
  const float* bk   = (const float*)d_in[7];
  const float* Wv   = (const float*)d_in[8];
  const float* bv   = (const float*)d_in[9];
  const float* Wo   = (const float*)d_in[10];
  const float* bo   = (const float*)d_in[11];
  const float* rel  = (const float*)d_in[12];
  const float* We   = (const float*)d_in[13];

  char*  ws  = (char*)d_ws;
  size_t off = 0;
  auto carve = [&](size_t bytes) -> char* {
    char* p = ws + off;
    off = (off + bytes + 255) & ~(size_t)255;
    return p;
  };
  const size_t BPACK_SZ = (size_t)NTILES * KTILES * 32 * 16 * 2; // 2 MB
  unsigned short* xb    = (unsigned short*)carve((size_t)S_PAD * DMODEL * 2);
  unsigned short* attnb = (unsigned short*)carve((size_t)S_PAD * DMODEL * 2);
  unsigned short* bpQ   = (unsigned short*)carve(BPACK_SZ);
  unsigned short* bpK   = (unsigned short*)carve(BPACK_SZ);
  unsigned short* bpV   = (unsigned short*)carve(BPACK_SZ);
  unsigned short* bpO   = (unsigned short*)carve(BPACK_SZ);
  unsigned short* qm    = (unsigned short*)carve((size_t)NHEAD * S_PAD * HDIM * 2);
  unsigned short* km    = (unsigned short*)carve((size_t)NHEAD * S_PAD * HDIM * 2);
  unsigned short* vtm   = (unsigned short*)carve((size_t)NHEAD * HDIM * S_PAD * 2);
  float*          posT  = (float*)carve((size_t)NHEAD * RELN * 4);
  float*          ev    = (float*)carve((size_t)NHEAD * NEDGE * 4);

  // 1. pack inputs / tables
  pack_x_kernel<<<(S_PAD * DMODEL) / 256, 256, 0, stream>>>(x, xb);
  pack_pos_kernel<<<(NHEAD * RELN + 255) / 256, 256, 0, stream>>>(rel, posT);
  pack_w_kernel<<<dim3(256, 4), 256, 0, stream>>>(Wq, Wk, Wv, Wo,
                                                  bpQ, bpK, bpV, bpO);
  edge_kernel<<<(NHEAD * NEDGE * 32 + 255) / 256, 256, 0, stream>>>(te, We, ev);

  // 2. Q/K/V projections (WMMA bf16)
  gemm_bf16_kernel<<<dim3(MTILES, 4), 128, 0, stream>>>(xb, bpQ, bq, qm, 0);
  gemm_bf16_kernel<<<dim3(MTILES, 4), 128, 0, stream>>>(xb, bpK, bk, km, 1);
  gemm_bf16_kernel<<<dim3(MTILES, 4), 128, 0, stream>>>(xb, bpV, bv, vtm, 2);

  // 3. flash attention (WMMA bf16, streaming softmax)
  attention_kernel<<<dim3(NHEAD, MTILES / 4), 128, 0, stream>>>(
      qm, km, vtm, posT, ev, mask, attnb);

  // 4. output projection (WMMA bf16, f32 out + bias)
  gemm_bf16_kernel<<<dim3(MTILES, 4), 128, 0, stream>>>(attnb, bpO, bo, d_out, 3);
}
